// SpeechAttention_27865747817305
// MI455X (gfx1250) — matrix operand order repaired and checked by reading, not compiled, not run on hardware
//
#include <hip/hip_runtime.h>

typedef __bf16 bf16_t;
typedef __attribute__((ext_vector_type(16))) __bf16 v16bf;
typedef __attribute__((ext_vector_type(8)))  __bf16 v8bf;
typedef __attribute__((ext_vector_type(8)))  float  v8f;
typedef int v4i_ __attribute__((vector_size(16)));

#define B_     2
#define CH     512
#define MIDC   512
#define GRP    8
#define PATCH_ 16
#define W_     512           // wordNum
#define D_     8192          // MID*PATCH
#define T_     8192
#define ICG    64            // in-channels per group
#define OCG    192           // out-channels per group (1536/8)
#define KH     5
#define KCONV  (ICG * KH)    // 320 = im2col K
#define NEGV   (-1000000000.0f)

#if defined(__has_builtin)
#if __has_builtin(__builtin_amdgcn_global_load_async_to_lds_b128) && \
    __has_builtin(__builtin_amdgcn_s_wait_asynccnt)
#define HAVE_ASYNC_LDS 1
#endif
#endif
#ifndef HAVE_ASYNC_LDS
#define HAVE_ASYNC_LDS 0
#endif

static constexpr size_t BWD = (size_t)B_ * W_ * D_;        // 8,388,608
// workspace byte offsets (all 32B-aligned)
static constexpr size_t OFF_QRAW = 0;
static constexpr size_t OFF_KRAW = OFF_QRAW + BWD * 4;
static constexpr size_t OFF_VRAW = OFF_KRAW + BWD * 4;
static constexpr size_t OFF_QN   = OFF_VRAW + BWD * 4;
static constexpr size_t OFF_KN   = OFF_QN   + BWD * 2;
static constexpr size_t OFF_VT   = OFF_KN   + BWD * 2;
static constexpr size_t OFF_SBF  = OFF_VT   + BWD * 2;
static constexpr size_t OFF_ATT  = OFF_SBF  + (size_t)B_ * W_ * W_ * 2;
static constexpr size_t ATT_ROWS = (size_t)T_ + 2;          // halo rows t=-1, t=T
static constexpr size_t OFF_WPK  = OFF_ATT + (size_t)B_ * ATT_ROWS * MIDC * 2;
static constexpr size_t OFF_WQP  = OFF_WPK + (size_t)3 * MIDC * MIDC * 2;

// ---------------------------------------------------------------------------
// WMMA operand helpers (this code's k order, not ISA table order): A lanes hold M=lane&15 with 16 contiguous
// K at k+half*16 (matching B); B lanes hold 16 contiguous K.
// ---------------------------------------------------------------------------
__device__ __forceinline__ v16bf load_a16(const bf16_t* row, int k, int half)
{
    const v8bf* p = (const v8bf*)(row + k + half * 16);
    v8bf lo = p[0];
    v8bf hi = p[1];
    return __builtin_shufflevector(lo, hi, 0, 1, 2, 3, 4, 5, 6, 7,
                                           8, 9, 10, 11, 12, 13, 14, 15);
}

// Stage a 64-row x 512-col bf16 panel (64KB) into LDS, shared by all 8 waves.
// Uses GLOBAL_LOAD_ASYNC_TO_LDS_B128 (ASYNCcnt) when available.
__device__ __forceinline__ void stage_panel(const bf16_t* __restrict__ gsrc,
                                            bf16_t* lds, int tid)
{
    #pragma unroll 4
    for (int i = 0; i < 16; ++i) {
        int e = tid + 256 * i;              // 4096 x 16B chunks
#if HAVE_ASYNC_LDS
        __builtin_amdgcn_global_load_async_to_lds_b128(
            (__attribute__((address_space(1))) v4i_*)(gsrc + e * 8),
            (__attribute__((address_space(3))) v4i_*)(lds + e * 8), 0, 0);
#else
        *(v8bf*)(lds + e * 8) = *(const v8bf*)(gsrc + e * 8);
#endif
    }
#if HAVE_ASYNC_LDS
    __builtin_amdgcn_s_wait_asynccnt(0);
#endif
    __syncthreads();
}

// ---------------------------------------------------------------------------
// 0a) Pack W_qkv (1536,64,5) fp32 -> bf16; layout is already oc x (ic*5+kh)
// ---------------------------------------------------------------------------
__global__ void pack_wqkv_kernel(const float* __restrict__ Wq,
                                 bf16_t* __restrict__ wqp)
{
    int idx = blockIdx.x * 256 + threadIdx.x;   // 1536*320 = 491520
    wqp[idx] = (bf16_t)Wq[idx];
}

// ---------------------------------------------------------------------------
// 0b) Pack W_out (oc,ic,3) -> 3 bf16 matrices Wk[kk][oc][ic]
// ---------------------------------------------------------------------------
__global__ void pack_wout_kernel(const float* __restrict__ Wo,
                                 bf16_t* __restrict__ wpk)
{
    int idx = blockIdx.x * 256 + threadIdx.x;   // 3*512*512
    int ic = idx & 511;
    int oc = (idx >> 9) & 511;
    int kk = idx >> 18;
    wpk[idx] = (bf16_t)Wo[((size_t)oc * MIDC + ic) * 3 + kk];
}

__global__ void zero_att_pad_kernel(bf16_t* __restrict__ attT)
{
    int tid = blockIdx.x * 256 + threadIdx.x;   // 2048
    int mid = tid & 511;
    int r   = (tid >> 9) & 1;
    int b   = tid >> 10;
    size_t row = r ? (ATT_ROWS - 1) : 0;
    attT[(size_t)b * ATT_ROWS * MIDC + row * MIDC + mid] = (bf16_t)0.0f;
}

// ---------------------------------------------------------------------------
// 1) Grouped conv(k=5) as WMMA GEMM: per (b,group,4 words) block, build the
//    im2col tile N-major in LDS (64 x 320 bf16 = 40KB), then each of 8 waves
//    runs a 96x16 strip: MT=6 WMMA tiles sharing one LDS B-fragment/step.
//    Bias + drop-mask (k slice) in the fp32 epilogue.
// ---------------------------------------------------------------------------
__global__ void conv_qkv_wmma_kernel(const float* __restrict__ x,
                                     const bf16_t* __restrict__ wqp,
                                     const float* __restrict__ bq,
                                     const float* __restrict__ mask,
                                     float* __restrict__ qraw,
                                     float* __restrict__ kraw,
                                     float* __restrict__ vraw)
{
    __shared__ __align__(32) bf16_t Bs[64][KCONV];   // [n][kidx], 40KB
    const int tid = threadIdx.x;
    const int blk = blockIdx.x;             // B_*GRP*(W_/4) = 2048
    const int wc  = blk & 127;
    const int g   = (blk >> 7) & 7;
    const int b   = blk >> 10;
    const int w0  = wc * 4;

    // im2col fill: Bs[n][ic*5+kh] = x[b, g*64+ic, w0 + (n>>4) + kh - 2, n&15]
    for (int e = tid; e < 64 * KCONV; e += 256) {
        int n    = e / KCONV;
        int kidx = e - n * KCONV;
        int ic   = kidx / KH;
        int kh   = kidx - ic * KH;
        int wl   = n >> 4, p = n & 15;
        int w    = w0 + wl + kh - 2;
        float v  = 0.0f;
        if (w >= 0 && w < W_)
            v = x[(((size_t)b * CH + g * ICG + ic) * W_ + w) * PATCH_ + p];
        Bs[n][kidx] = (bf16_t)v;
    }
    __syncthreads();

    const int wave = tid >> 5;
    const int lane = tid & 31;
    const int half = lane >> 4, l16 = lane & 15;
    const int nj   = wave & 3;              // n-tile (16 cols)
    const int mblk = wave >> 2;             // 0/1 -> m-tiles mblk*6 .. +5
    const int n    = nj * 16 + l16;

    v8f acc[6] = {};
    for (int k = 0; k < KCONV; k += 32) {
        v16bf bb = *(const v16bf*)(&Bs[n][k + half * 16]);
        #pragma unroll
        for (int i = 0; i < 6; ++i) {
            const bf16_t* arow =
                wqp + (size_t)(g * OCG + (mblk * 6 + i) * 16 + l16) * KCONV;
            v16bf a = load_a16(arow, k, half);
            acc[i] = __builtin_amdgcn_wmma_f32_16x16x32_bf16(false, a, false, bb,
                                                             (short)0, acc[i],
                                                             false, false);
        }
    }
    const int wl = n >> 4, p = n & 15;
    const int w  = w0 + wl;
    #pragma unroll
    for (int i = 0; i < 6; ++i) {
        #pragma unroll
        for (int r = 0; r < 8; ++r) {
            int c = g * OCG + (mblk * 6 + i) * 16 + half * 8 + r;
            float v = acc[i][r] + bq[c];
            int qi  = c >> 9;
            int mid = c & 511;
            if (qi == 1)
                v += mask[(((size_t)b * MIDC + mid) * W_ + w) * PATCH_ + p] * NEGV;
            float* dst = (qi == 0) ? qraw : (qi == 1 ? kraw : vraw);
            dst[((size_t)b * W_ + w) * D_ + mid * PATCH_ + p] = v;
        }
    }
}

// ---------------------------------------------------------------------------
// 2) Row softmax of q over d=8192 -> bf16. One block per (b,w) row.
// ---------------------------------------------------------------------------
__global__ void softmax_q_kernel(const float* __restrict__ qraw,
                                 bf16_t* __restrict__ qn)
{
    __shared__ float red[256];
    const int tid = threadIdx.x;
    const size_t row = blockIdx.x;          // B_*W_ rows
    const float* r = qraw + row * D_;
    float m = -3.4e38f;
    for (int i = tid; i < D_; i += 256) m = fmaxf(m, r[i]);
    red[tid] = m; __syncthreads();
    for (int s = 128; s > 0; s >>= 1) {
        if (tid < s) red[tid] = fmaxf(red[tid], red[tid + s]);
        __syncthreads();
    }
    m = red[0]; __syncthreads();
    float sum = 0.0f;
    for (int i = tid; i < D_; i += 256) sum += __expf(r[i] - m);
    red[tid] = sum; __syncthreads();
    for (int s = 128; s > 0; s >>= 1) {
        if (tid < s) red[tid] += red[tid + s];
        __syncthreads();
    }
    const float inv = 1.0f / red[0];
    bf16_t* o = qn + row * D_;
    for (int i = tid; i < D_; i += 256)
        o[i] = (bf16_t)(__expf(r[i] - m) * inv);
}

// ---------------------------------------------------------------------------
// 3) Column softmax of k over W=512 (online max/sum) -> bf16
// ---------------------------------------------------------------------------
__global__ void softmax_k_kernel(const float* __restrict__ kraw,
                                 bf16_t* __restrict__ kn)
{
    const int col = blockIdx.x * 256 + threadIdx.x;   // B_*D_ = 16384
    const int d = col & (D_ - 1);
    const int b = col >> 13;
    const float* base = kraw + (size_t)b * W_ * D_ + d;
    float m = -3.4e38f, s = 0.0f;
    for (int w = 0; w < W_; ++w) {
        float v  = base[(size_t)w * D_];
        float nm = fmaxf(m, v);
        s = s * __expf(m - nm) + __expf(v - nm);
        m = nm;
    }
    const float inv = 1.0f / s;
    bf16_t* ob = kn + (size_t)b * W_ * D_ + d;
    for (int w = 0; w < W_; ++w)
        ob[(size_t)w * D_] = (bf16_t)(__expf(base[(size_t)w * D_] - m) * inv);
}

// ---------------------------------------------------------------------------
// 4) v -> bf16, transposed to (b, d, W): WMMA B-operand K-loads contiguous
// ---------------------------------------------------------------------------
__global__ void transpose_v_kernel(const float* __restrict__ vraw,
                                   bf16_t* __restrict__ vt)
{
    size_t idx = (size_t)blockIdx.x * 256 + threadIdx.x;   // BWD
    int w = (int)(idx & (W_ - 1));
    size_t r = idx >> 9;
    int d = (int)(r & (D_ - 1));
    int b = (int)(r >> 13);
    vt[idx] = (bf16_t)vraw[((size_t)b * W_ + w) * D_ + d];
}

// ---------------------------------------------------------------------------
// 5) s = softmax(q) @ softmax(k)^T  (M=N=512, K=8192), 2x2 register blocking
// ---------------------------------------------------------------------------
__global__ void gemm_qk_kernel(const bf16_t* __restrict__ qn,
                               const bf16_t* __restrict__ kn,
                               bf16_t* __restrict__ sbf)
{
    const int wave = threadIdx.x >> 5;
    const int lane = threadIdx.x & 31;
    const int half = lane >> 4, l16 = lane & 15;
    const int tile = blockIdx.x * 8 + wave;     // B_*16*16 groups
    const int b  = tile >> 8;
    const int t  = tile & 255;
    const int mg = t >> 4, ng = t & 15;
    const bf16_t* arow[2];
    const bf16_t* brow[2];
    #pragma unroll
    for (int i = 0; i < 2; ++i) {
        arow[i] = qn + ((size_t)b * W_ + mg * 32 + i * 16 + l16) * D_;
        brow[i] = kn + ((size_t)b * W_ + ng * 32 + i * 16 + l16) * D_ + half * 16;
    }
    v8f acc[2][2] = {};
    for (int k = 0; k < D_; k += 32) {
        __builtin_prefetch(arow[0] + k + 512, 0, 3);
        __builtin_prefetch(arow[1] + k + 512, 0, 3);
        __builtin_prefetch(brow[0] + k + 512, 0, 3);
        __builtin_prefetch(brow[1] + k + 512, 0, 3);
        v16bf a[2], bb[2];
        #pragma unroll
        for (int i = 0; i < 2; ++i) a[i]  = load_a16(arow[i], k, half);
        #pragma unroll
        for (int j = 0; j < 2; ++j) bb[j] = *(const v16bf*)(brow[j] + k);
        #pragma unroll
        for (int i = 0; i < 2; ++i)
            #pragma unroll
            for (int j = 0; j < 2; ++j)
                acc[i][j] = __builtin_amdgcn_wmma_f32_16x16x32_bf16(
                    false, a[i], false, bb[j], (short)0, acc[i][j], false, false);
    }
    bf16_t* o = sbf + (size_t)b * W_ * W_;
    #pragma unroll
    for (int i = 0; i < 2; ++i)
        #pragma unroll
        for (int j = 0; j < 2; ++j)
            #pragma unroll
            for (int r = 0; r < 8; ++r) {
                int m = mg * 32 + i * 16 + half * 8 + r;
                int n = ng * 32 + j * 16 + l16;
                o[(size_t)m * W_ + n] = (bf16_t)acc[i][j][r];
            }
}

// ---------------------------------------------------------------------------
// 6) attn = s @ v  (M=512, N=8192, K=512), 4x4 blocking (64x64 per wave).
//    The 64x512 A-panel (shared by all 8 waves: same mg) is staged once into
//    LDS via async global->LDS copies; A-fragments come from ds_load.
//    Scatter-store pre-transposed (t, mid) with zero halo rows.
// ---------------------------------------------------------------------------
__global__ void gemm_sv_kernel(const bf16_t* __restrict__ sbf,
                               const bf16_t* __restrict__ vt,
                               bf16_t* __restrict__ attT)
{
    __shared__ __align__(32) bf16_t As[64 * W_];    // 64KB
    const int tid  = threadIdx.x;
    const int wave = tid >> 5;
    const int lane = tid & 31;
    const int half = lane >> 4, l16 = lane & 15;
    const int tileBase = blockIdx.x * 8;            // B_*8*128 groups
    const int b  = tileBase >> 10;                  // uniform across block
    const int mg = (tileBase & 1023) >> 7;          // uniform across block
    const int ng = (tileBase & 127) + wave;

    stage_panel(sbf + (size_t)b * W_ * W_ + (size_t)(mg * 64) * W_, As, tid);

    const bf16_t* brow[4];
    #pragma unroll
    for (int j = 0; j < 4; ++j)
        brow[j] = vt + (size_t)b * D_ * W_
                     + (size_t)(ng * 64 + j * 16 + l16) * W_ + half * 16;
    v8f acc[4][4] = {};
    for (int k = 0; k < W_; k += 32) {
        v16bf a[4], bb[4];
        #pragma unroll
        for (int i = 0; i < 4; ++i)
            a[i] = load_a16(As + (size_t)(i * 16 + l16) * W_, k, half);
        #pragma unroll
        for (int j = 0; j < 4; ++j) bb[j] = *(const v16bf*)(brow[j] + k);
        #pragma unroll
        for (int i = 0; i < 4; ++i)
            #pragma unroll
            for (int j = 0; j < 4; ++j)
                acc[i][j] = __builtin_amdgcn_wmma_f32_16x16x32_bf16(
                    false, a[i], false, bb[j], (short)0, acc[i][j], false, false);
    }
    bf16_t* ob = attT + (size_t)b * ATT_ROWS * MIDC;
    #pragma unroll
    for (int j = 0; j < 4; ++j) {
        int n   = ng * 64 + j * 16 + l16;    // d
        int p   = n & 15;
        int mid = n >> 4;
        #pragma unroll
        for (int i = 0; i < 4; ++i)
            #pragma unroll
            for (int r = 0; r < 8; ++r) {
                int w  = mg * 64 + i * 16 + half * 8 + r;
                int tt = w * 16 + p;
                ob[(size_t)(tt + 1) * MIDC + mid] = (bf16_t)acc[i][j][r];
            }
    }
}

// ---------------------------------------------------------------------------
// 7) Final conv1d(k=3): 3 shifted WMMA GEMMs, 4x4 blocking; per-kk A-panel
//    async-staged into LDS (reused buffer), fp32 out + bias.
// ---------------------------------------------------------------------------
__global__ void gemm_out_kernel(const bf16_t* __restrict__ wpk,
                                const bf16_t* __restrict__ attT,
                                const float* __restrict__ bout,
                                float* __restrict__ out)
{
    __shared__ __align__(32) bf16_t As[64 * MIDC];  // 64KB
    const int tid  = threadIdx.x;
    const int wave = tid >> 5;
    const int lane = tid & 31;
    const int half = lane >> 4, l16 = lane & 15;
    const int tileBase = blockIdx.x * 8;            // B_*8*128 groups
    const int b  = tileBase >> 10;
    const int mg = (tileBase & 1023) >> 7;
    const int ng = (tileBase & 127) + wave;

    v8f acc[4][4] = {};
    for (int kk = 0; kk < 3; ++kk) {
        stage_panel(wpk + (size_t)kk * MIDC * MIDC + (size_t)(mg * 64) * MIDC,
                    As, tid);
        const bf16_t* brow[4];
        #pragma unroll
        for (int j = 0; j < 4; ++j)
            brow[j] = attT + (size_t)b * ATT_ROWS * MIDC
                           + (size_t)(ng * 64 + j * 16 + l16 + kk) * MIDC
                           + half * 16;
        for (int k = 0; k < MIDC; k += 32) {
            v16bf a[4], bb[4];
            #pragma unroll
            for (int i = 0; i < 4; ++i)
                a[i] = load_a16(As + (size_t)(i * 16 + l16) * MIDC, k, half);
            #pragma unroll
            for (int j = 0; j < 4; ++j) bb[j] = *(const v16bf*)(brow[j] + k);
            #pragma unroll
            for (int i = 0; i < 4; ++i)
                #pragma unroll
                for (int j = 0; j < 4; ++j)
                    acc[i][j] = __builtin_amdgcn_wmma_f32_16x16x32_bf16(
                        false, a[i], false, bb[j], (short)0, acc[i][j],
                        false, false);
        }
        __syncthreads();   // As reused next kk
    }
    #pragma unroll
    for (int i = 0; i < 4; ++i)
        #pragma unroll
        for (int r = 0; r < 8; ++r) {
            int oc = mg * 64 + i * 16 + half * 8 + r;
            float bias = bout[oc];
            #pragma unroll
            for (int j = 0; j < 4; ++j) {
                int n = ng * 64 + j * 16 + l16;
                out[((size_t)b * CH + oc) * T_ + n] = acc[i][j][r] + bias;
            }
        }
}

// ---------------------------------------------------------------------------
extern "C" void kernel_launch(void* const* d_in, const int* in_sizes, int n_in,
                              void* d_out, int out_size, void* d_ws, size_t ws_size,
                              hipStream_t stream)
{
    (void)in_sizes; (void)n_in; (void)out_size; (void)ws_size;
    const float* x    = (const float*)d_in[0];
    const float* Wqkv = (const float*)d_in[1];
    const float* bqkv = (const float*)d_in[2];
    const float* Wout = (const float*)d_in[3];
    const float* bout = (const float*)d_in[4];
    const float* mask = (const float*)d_in[5];
    float* out = (float*)d_out;

    char* ws = (char*)d_ws;
    float*  qraw = (float*)(ws + OFF_QRAW);
    float*  kraw = (float*)(ws + OFF_KRAW);
    float*  vraw = (float*)(ws + OFF_VRAW);
    bf16_t* qn   = (bf16_t*)(ws + OFF_QN);
    bf16_t* kn   = (bf16_t*)(ws + OFF_KN);
    bf16_t* vt   = (bf16_t*)(ws + OFF_VT);
    bf16_t* sbf  = (bf16_t*)(ws + OFF_SBF);
    bf16_t* attT = (bf16_t*)(ws + OFF_ATT);
    bf16_t* wpk  = (bf16_t*)(ws + OFF_WPK);
    bf16_t* wqp  = (bf16_t*)(ws + OFF_WQP);

    pack_wqkv_kernel<<<(3 * MIDC * KCONV) / 256, 256, 0, stream>>>(Wqkv, wqp);
    pack_wout_kernel<<<(3 * MIDC * MIDC) / 256, 256, 0, stream>>>(Wout, wpk);
    zero_att_pad_kernel<<<2048 / 256, 256, 0, stream>>>(attT);
    conv_qkv_wmma_kernel<<<B_ * GRP * (W_ / 4), 256, 0, stream>>>(
        x, wqp, bqkv, mask, qraw, kraw, vraw);
    softmax_q_kernel<<<B_ * W_, 256, 0, stream>>>(qraw, qn);
    softmax_k_kernel<<<(B_ * D_) / 256, 256, 0, stream>>>(kraw, kn);
    transpose_v_kernel<<<(int)(BWD / 256), 256, 0, stream>>>(vraw, vt);
    gemm_qk_kernel<<<(B_ * 16 * 16) / 8, 256, 0, stream>>>(qn, kn, sbf);
    gemm_sv_kernel<<<(B_ * 8 * 128) / 8, 256, 0, stream>>>(sbf, vt, attT);
    gemm_out_kernel<<<(B_ * 8 * 128) / 8, 256, 0, stream>>>(wpk, attT, bout, out);
}
